// CSCAModule_71949292142813
// MI455X (gfx1250) — matrix operand order repaired and checked by reading, not compiled, not run on hardware
//
#include <hip/hip_runtime.h>

// ---------------------------------------------------------------------------
// CSCA module for MI455X (gfx1250).  Heavy GEMMs use v_wmma_f32_16x16x32_bf16
// (fp32 operands -> bf16 at staging, fp32 accumulate).  wave32, 256-thread
// blocks (8 waves).  LDS operand panels are stored PRE-SWIZZLED in per-lane
// fragment order so each WMMA operand load is one 32-byte LDS vector read.
// ---------------------------------------------------------------------------

#define NB 24          // batch (both support and query)
#define DD 512         // channels
#define HW 25          // spatial tokens (5*5)
#define SZ (NB*DD*HW)  // 307200 elements per (b,C,HW) tensor

// workspace offsets (in floats)
#define OFF_QAC 0
#define OFF_VAC (SZ)
#define OFF_KBC (2*SZ)
#define OFF_VBC (3*SZ)
#define OFF_QAP (4*SZ)
#define OFF_VAP (5*SZ)
#define OFF_KBP (6*SZ)
#define OFF_VBP (7*SZ)
#define OFF_DUMP (8*SZ)   // sink for unused qkv outputs (ka_p / qb_p / ka_c / qb_c)
#define OFF_ATTP (9*SZ)   // att_p mean: 24*25*25
#define OFF_OUTP (10*SZ)  // out_p: 24*25*512  (q,n,c) row-major
#define OFF_OUTC (11*SZ)  // out_c (pre fc_o): 24*512*25

typedef unsigned short u16t;
typedef __attribute__((ext_vector_type(16))) __bf16 v16bf;
typedef __attribute__((ext_vector_type(16))) unsigned short v16us;
typedef __attribute__((ext_vector_type(8)))  float v8f;
typedef __attribute__((ext_vector_type(8)))  unsigned int v8u;

__device__ __forceinline__ u16t f2bf(float f) {
  unsigned x = __builtin_bit_cast(unsigned, f);
  x += 0x7FFFu + ((x >> 16) & 1u);   // round-to-nearest-even
  return (u16t)(x >> 16);
}

// Per-lane K pattern of the 16-bit A fragment (ISA 7.12.2, 16x32):
// lane l: row m = l&15, hi = l>>4; VGPR v holds K = (v>=4?16:0)+hi*8+2*(v&3).
__device__ __forceinline__ int a_kbase(int v, int hi) {
  return ((v & 4) << 2) + hi * 8 + ((v & 3) << 1);
}

// Load one fragment (16 bf16) from a pre-swizzled LDS slot: one 32B read.
__device__ __forceinline__ v16bf frag_load(const u16t* p) {
  return __builtin_bit_cast(v16bf, *(const v8u*)p);
}

// A fragment gathered from global fp32 where A(m,k) = src[k*ld + m]
// (A = transpose of row-major src). Zero-pads rows m >= mlim. Scale folded in.
__device__ __forceinline__ v16bf frag_a_gT(const float* src, int ld, int m0,
                                           int mlim, int lane, float scale) {
  int m = m0 + (lane & 15), hi = lane >> 4;
  bool ok = (m < mlim);
  v16us u;
#pragma unroll
  for (int v = 0; v < 8; ++v) {
    int kb = a_kbase(v, hi);
    u[2 * v]     = ok ? f2bf(src[kb * ld + m] * scale)       : (u16t)0;
    u[2 * v + 1] = ok ? f2bf(src[(kb + 1) * ld + m] * scale) : (u16t)0;
  }
  return __builtin_bit_cast(v16bf, u);
}

// A fragment gathered from global fp32 where A(m,k) = src[m*ld + k],
// zero-padding columns k >= klim. Scale folded in.
__device__ __forceinline__ v16bf frag_a_g(const float* src, int ld, int klim,
                                          int lane, float scale) {
  int m = lane & 15, hi = lane >> 4;
  v16us u;
#pragma unroll
  for (int v = 0; v < 8; ++v) {
    int kb = a_kbase(v, hi);
    u[2 * v]     = (kb < klim)     ? f2bf(src[m * ld + kb] * scale)     : (u16t)0;
    u[2 * v + 1] = (kb + 1 < klim) ? f2bf(src[m * ld + kb + 1] * scale) : (u16t)0;
  }
  return __builtin_bit_cast(v16bf, u);
}

__device__ __forceinline__ v8f wmma_bf16(v16bf a, v16bf b, v8f c) {
  return __builtin_amdgcn_wmma_f32_16x16x32_bf16(false, a, false, b, (short)0,
                                                 c, false, false);
}

// Swizzled-slot index helpers (u16 units). Slot layout: [tile][lane 0..31][16].
// B element (local k 0..31, local n 0..15): lane = n + 16*((k>>3)&1), idx = (k&7) + 8*(k>>4).
// A element (m 0..15, local k 0..31): lane = m + 16*((k>>3)&1),
//                                     idx  = (k&7) + 8*(k>>4).

// ---------------------------------------------------------------------------
// K1: channel-branch qkv.  t[b,c,m] = sum_n x[b,c,n] * Wc[m,n], m in [0,75).
// grid: 48 blocks = {support,query} x 24 batches.  (46 MFLOP -> VALU is fine.)
// ---------------------------------------------------------------------------
__global__ void __launch_bounds__(256)
qkvc_kernel(const float* __restrict__ support, const float* __restrict__ query,
            const float* __restrict__ Wc, float* __restrict__ ws) {
  __shared__ float w[75 * HW];
  int blk = blockIdx.x;
  int isQ = blk / NB, b = blk % NB;
  const float* x = (isQ ? query : support) + (size_t)b * DD * HW;
  float* dq = ws + (isQ ? OFF_DUMP : OFF_QAC);
  float* dk = ws + (isQ ? OFF_KBC : OFF_DUMP);
  float* dv = ws + (isQ ? OFF_VBC : OFF_VAC);
  for (int i = threadIdx.x; i < 75 * HW; i += 256) w[i] = Wc[i];
  __syncthreads();
  for (int i = threadIdx.x; i < DD * 75; i += 256) {
    int c = i / 75, m = i % 75;
    const float* xr = x + c * HW;
    float acc = 0.f;
#pragma unroll
    for (int n = 0; n < HW; ++n) acc += xr[n] * w[m * HW + n];
    float* dst = (m < 25) ? dq : (m < 50) ? dk : dv;
    dst[b * DD * HW + c * HW + (m % 25)] = acc;
  }
}

// ---------------------------------------------------------------------------
// K2 (WMMA): position-branch qkv.  t[b,n,d] = sum_c x[b,c,n]*Wp[d,c].
// A(25x512,pad32) gathered from global; B staged pre-swizzled per 32-K panel.
// Result scattered through the raw (b,25,1536)->(b,512,3,25) reinterpretation.
// grid: 48 batches * 12 d-blocks = 576 blocks.
// ---------------------------------------------------------------------------
__global__ void __launch_bounds__(256)
qkvp_kernel(const float* __restrict__ support, const float* __restrict__ query,
            const float* __restrict__ Wp, float* __restrict__ ws) {
  __shared__ __align__(32) u16t Bs[8 * 32 * 16];   // 8 n-tiles, swizzled (8KB)
  int blk = blockIdx.x;
  int dblk = blk % 12, bi = blk / 12;
  int isQ = bi / NB, b = bi % NB;
  const float* x = (isQ ? query : support) + (size_t)b * DD * HW;
  int lane = threadIdx.x & 31, wave = threadIdx.x >> 5;
  int mt = wave & 1, nt0 = wave >> 1;  // 2 M-tiles x 8 N-tiles over 8 waves
  v8f acc0 = {}, acc1 = {};
  int d0 = dblk * 128;
  for (int kc = 0; kc < DD; kc += 32) {
    __syncthreads();
    for (int i = threadIdx.x; i < 8 * 32 * 8; i += 256) {  // u32 slots
      int tile = i >> 8, l2 = (i >> 3) & 31, t2 = i & 7;
    int k0 = (l2 >> 4) * 8 + ((2 * t2) & 7) + 16 * (t2 >> 2);
      int e = d0 + tile * 16 + (l2 & 15);
      unsigned lo = f2bf(Wp[(size_t)e * DD + kc + k0]);
      unsigned hi = f2bf(Wp[(size_t)e * DD + kc + k0 + 1]);
      ((unsigned*)Bs)[i] = lo | (hi << 16);
    }
    __syncthreads();
    v16bf a  = frag_a_gT(x + kc * HW, HW, mt * 16, HW, lane, 1.0f);
    v16bf b0 = frag_load(&Bs[(nt0 * 32 + lane) * 16]);
    v16bf b1 = frag_load(&Bs[((nt0 + 4) * 32 + lane) * 16]);
    acc0 = wmma_bf16(a, b0, acc0);
    acc1 = wmma_bf16(a, b1, acc1);
  }
  float* dq = ws + (isQ ? OFF_DUMP : OFF_QAP);
  float* dk = ws + (isQ ? OFF_KBP : OFF_DUMP);
  float* dv = ws + (isQ ? OFF_VBP : OFF_VAP);
  int hi = lane >> 4, nloc = lane & 15;
#pragma unroll
  for (int t = 0; t < 2; ++t) {
    v8f acc = t ? acc1 : acc0;
    int nt = nt0 + t * 4;
#pragma unroll
    for (int v = 0; v < 8; ++v) {
      int m = mt * 16 + hi * 8 + v;  // spatial index n
      if (m < HW) {
        int d = d0 + nt * 16 + nloc;
        int f = m * 1536 + d;        // raw flat reinterpretation
        int c = f / 75, r = f % 75;
        float* dst = (r < 25) ? dq : (r < 50) ? dk : dv;
        dst[b * DD * HW + c * HW + (r % 25)] = acc[v];
      }
    }
  }
}

// ---------------------------------------------------------------------------
// K3 (WMMA): position attention scores + softmax + mean over s.
// att[q,n,m] = mean_s softmax_m( sum_c qa_p[s,c,n]*kb_p[q,c,m] / sqrt(512) )
// B (K=512 x 32) staged pre-swizzled once per block. grid: 24 blocks.
// ---------------------------------------------------------------------------
__global__ void __launch_bounds__(256)
posatt_kernel(float* __restrict__ ws) {
  __shared__ __align__(32) u16t Bs[32 * 32 * 16];  // (kcChunk*2+nt) tiles, 32KB
  __shared__ float sc[32 * 32];
  __shared__ float accm[32 * 32];
  int q = blockIdx.x;
  const float* kb = ws + OFF_KBP + (size_t)q * DD * HW;
  for (int i = threadIdx.x; i < 32 * 32 * 8; i += 256) {   // u32 slots
    int tile = i >> 8, l2 = (i >> 3) & 31, t2 = i & 7;
    int kch = tile >> 1, nt = tile & 1;
    int k = kch * 32 + (l2 >> 4) * 8 + ((2 * t2) & 7) + 16 * (t2 >> 2);            // channel c (K dim)
    int m = nt * 16 + (l2 & 15);                           // spatial col
    unsigned lo = (m < HW) ? (unsigned)f2bf(kb[k * HW + m]) : 0u;
    unsigned hi = (m < HW) ? (unsigned)f2bf(kb[(k + 1) * HW + m]) : 0u;
    ((unsigned*)Bs)[i] = lo | (hi << 16);
  }
  for (int i = threadIdx.x; i < 32 * 32; i += 256) accm[i] = 0.f;
  int lane = threadIdx.x & 31, wave = threadIdx.x >> 5;
  int hi = lane >> 4, nloc = lane & 15;
  const float scale = 0.044194173824159216f;  // 1/sqrt(512), folded into A
  for (int s = 0; s < NB; ++s) {
    const float* qa = ws + OFF_QAP + (size_t)s * DD * HW;
    __syncthreads();  // prev softmax finished; Bs/accm ready on 1st iter
    if (wave < 4) {
      int mt = wave >> 1, nt = wave & 1;
      v8f acc = {};
      for (int kc = 0; kc < DD; kc += 32) {
        v16bf a  = frag_a_gT(qa + kc * HW, HW, mt * 16, HW, lane, scale);
        v16bf bf = frag_load(&Bs[((((kc >> 5) << 1) + nt) * 32 + lane) * 16]);
        acc = wmma_bf16(a, bf, acc);
      }
#pragma unroll
      for (int v = 0; v < 8; ++v)
        sc[(mt * 16 + hi * 8 + v) * 32 + nt * 16 + nloc] = acc[v];
    }
    __syncthreads();
    if (threadIdx.x < HW) {  // softmax over m, accumulate mean
      int r = threadIdx.x;
      float mx = -1e30f;
      for (int m = 0; m < HW; ++m) mx = fmaxf(mx, sc[r * 32 + m]);
      float sum = 0.f;
      for (int m = 0; m < HW; ++m) sum += __expf(sc[r * 32 + m] - mx);
      float inv = 1.f / sum;
      for (int m = 0; m < HW; ++m)
        accm[r * 32 + m] += __expf(sc[r * 32 + m] - mx) * inv;
    }
  }
  __syncthreads();
  float* attp = ws + OFF_ATTP + q * HW * HW;
  for (int i = threadIdx.x; i < HW * HW; i += 256) {
    int n = i / HW, m = i % HW;
    attp[i] = accm[n * 32 + m] * (1.f / 24.f);
  }
}

// ---------------------------------------------------------------------------
// K4 (WMMA): out_p[q,n,c] = sum_m att[q,n,m] * vb_p[q,c,m].  K=25 (pad 32).
// grid: 24 blocks.
// ---------------------------------------------------------------------------
__global__ void __launch_bounds__(256)
outp_kernel(float* __restrict__ ws) {
  __shared__ __align__(32) u16t As[2 * 32 * 16];   // att, swizzled A (2KB)
  __shared__ __align__(32) u16t Bs[32 * 32 * 16];  // V, swizzled B (32KB)
  int q = blockIdx.x;
  const float* attp = ws + OFF_ATTP + q * HW * HW;
  const float* vb = ws + OFF_VBP + (size_t)q * DD * HW;
  for (int i = threadIdx.x; i < 2 * 32 * 8; i += 256) {    // A u32 slots
    int tile = i >> 8, l2 = (i >> 3) & 31, t = i & 7;
    int row = tile * 16 + (l2 & 15);
    int ahi = l2 >> 4;
    int k0 = ((2 * t) & 7) + 8 * ahi + 16 * (t >> 2);
    unsigned lo = (row < HW && k0 < HW) ? (unsigned)f2bf(attp[row * HW + k0]) : 0u;
    unsigned hi = (row < HW && k0 + 1 < HW) ? (unsigned)f2bf(attp[row * HW + k0 + 1]) : 0u;
    ((unsigned*)As)[i] = lo | (hi << 16);
  }
  for (int i = threadIdx.x; i < 32 * 32 * 8; i += 256) {   // B u32 slots
    int tile = i >> 8, l2 = (i >> 3) & 31, t2 = i & 7;
    int k0 = (l2 >> 4) * 8 + ((2 * t2) & 7) + 16 * (t2 >> 2);                      // K = spatial m
    int c = tile * 16 + (l2 & 15);
    unsigned lo = (k0 < HW) ? (unsigned)f2bf(vb[c * HW + k0]) : 0u;
    unsigned hi = (k0 + 1 < HW) ? (unsigned)f2bf(vb[c * HW + k0 + 1]) : 0u;
    ((unsigned*)Bs)[i] = lo | (hi << 16);
  }
  __syncthreads();
  int lane = threadIdx.x & 31, wave = threadIdx.x >> 5;
  int hi = lane >> 4, nloc = lane & 15;
  int mt = wave & 1, ntb = wave >> 1;
  float* outp = ws + OFF_OUTP + (size_t)q * HW * DD;
  v16bf a = frag_load(&As[(mt * 32 + lane) * 16]);
#pragma unroll
  for (int j = 0; j < 8; ++j) {
    int nt = ntb + 4 * j;
    v16bf bf = frag_load(&Bs[(nt * 32 + lane) * 16]);
    v8f acc = {};
    acc = wmma_bf16(a, bf, acc);
#pragma unroll
    for (int v = 0; v < 8; ++v) {
      int n = mt * 16 + hi * 8 + v;
      if (n < HW) outp[n * DD + nt * 16 + nloc] = acc[v];
    }
  }
}

// ---------------------------------------------------------------------------
// K5 (WMMA, dominant 7.6 GFLOP): channel attention, flash-style.
// Per block: q, 16-row c-block.  B fragments (s-invariant) are loaded into
// registers ONCE.  Loop s: 16x512 scores (4 single-step WMMAs/wave) kept in
// registers; softmax via LDS row reductions; mean accumulated in registers.
// Then (acc/24) @ vb_c via WMMA (K=512).  grid: 24*32 = 768 blocks.
// ---------------------------------------------------------------------------
__global__ void __launch_bounds__(256)
chanatt_kernel(float* __restrict__ ws) {
  __shared__ __align__(32) u16t Bkv[32 * 32 * 16]; // QK B then AV B (32KB)
  __shared__ __align__(32) u16t A2[16 * 32 * 16];  // acc bf16, swizzled (16KB)
  __shared__ float red[16 * 128];
  __shared__ float rs1[16], rs2[16];
  int q = blockIdx.x >> 5;
  int c0 = (blockIdx.x & 31) * 16;
  const float* kbc = ws + OFF_KBC + (size_t)q * DD * HW;
  // stage QK B pre-swizzled: tile = e>>4 (N), K = spatial n (pad 25->32)
  for (int i = threadIdx.x; i < 32 * 32 * 8; i += 256) {
    int tile = i >> 8, l2 = (i >> 3) & 31, t2 = i & 7;
    int k0 = (l2 >> 4) * 8 + ((2 * t2) & 7) + 16 * (t2 >> 2);
    int e = tile * 16 + (l2 & 15);
    unsigned lo = (k0 < HW) ? (unsigned)f2bf(kbc[e * HW + k0]) : 0u;
    unsigned hi = (k0 + 1 < HW) ? (unsigned)f2bf(kbc[e * HW + k0 + 1]) : 0u;
    ((unsigned*)Bkv)[i] = lo | (hi << 16);
  }
  __syncthreads();
  int lane = threadIdx.x & 31, wave = threadIdx.x >> 5;
  int hi = lane >> 4, nloc = lane & 15;
  int cg = wave * 16 + nloc;        // 0..127 column-group id
  v16bf bfr[4];
#pragma unroll
  for (int j = 0; j < 4; ++j)       // hoisted: s-invariant B fragments
    bfr[j] = frag_load(&Bkv[((wave + 8 * j) * 32 + lane) * 16]);
  v8f accs[4] = {{}, {}, {}, {}};
  const float scale = 0.2f;         // 1/sqrt(25), folded into A staging
  for (int s = 0; s < NB; ++s) {
    const float* qac = ws + OFF_QAC + (size_t)s * DD * HW + (size_t)c0 * HW;
    v16bf a = frag_a_g(qac, HW, HW, lane, scale);  // A[c][n], rows = c-block
    v8f scv[4];
#pragma unroll
    for (int j = 0; j < 4; ++j) {
      v8f z = {};
      scv[j] = wmma_bf16(a, bfr[j], z);
    }
    // row max
#pragma unroll
    for (int v = 0; v < 8; ++v) {
      float m = fmaxf(fmaxf(scv[0][v], scv[1][v]), fmaxf(scv[2][v], scv[3][v]));
      red[(hi * 8 + v) * 128 + cg] = m;
    }
    __syncthreads();
    if (threadIdx.x < 16) {
      float m = -1e30f;
      for (int i2 = 0; i2 < 128; ++i2) m = fmaxf(m, red[threadIdx.x * 128 + i2]);
      rs1[threadIdx.x] = m;
    }
    __syncthreads();
    // exp in place + row sum
#pragma unroll
    for (int v = 0; v < 8; ++v) {
      int row = hi * 8 + v;
      float mx = rs1[row], sum = 0.f;
#pragma unroll
      for (int j = 0; j < 4; ++j) {
        float e = __expf(scv[j][v] - mx);
        scv[j][v] = e;
        sum += e;
      }
      red[row * 128 + cg] = sum;
    }
    __syncthreads();
    if (threadIdx.x < 16) {
      float sm = 0.f;
      for (int i2 = 0; i2 < 128; ++i2) sm += red[threadIdx.x * 128 + i2];
      rs2[threadIdx.x] = 1.f / sm;
    }
    __syncthreads();
#pragma unroll
    for (int v = 0; v < 8; ++v) {
      float inv = rs2[hi * 8 + v];
#pragma unroll
      for (int j = 0; j < 4; ++j) accs[j][v] += scv[j][v] * inv;
    }
  }
  __syncthreads();  // all waves out of the s loop
  // stash acc/24 as bf16 into swizzled A slots (chunk = K/32)
#pragma unroll
  for (int j = 0; j < 4; ++j) {
    int col = (wave + 8 * j) * 16 + nloc;           // K index of AV GEMM
    int chunk = col >> 5, klocal = col & 31;
    int idx = (klocal & 7) + 8 * (klocal >> 4);
    int lsel = 16 * ((klocal >> 3) & 1);
#pragma unroll
    for (int v = 0; v < 8; ++v) {
      int row = hi * 8 + v;
      A2[((chunk * 32 + row + lsel) * 16) + idx] = f2bf(accs[j][v] * (1.f / 24.f));
    }
  }
  // restage Bkv as V (pre-swizzled): tile = (K/32)*2 + nt, K = channel e
  const float* vbc = ws + OFF_VBC + (size_t)q * DD * HW;
  for (int i = threadIdx.x; i < 32 * 32 * 8; i += 256) {
    int tile = i >> 8, l2 = (i >> 3) & 31, t2 = i & 7;
    int kch = tile >> 1, nt = tile & 1;
    int k = kch * 32 + (l2 >> 4) * 8 + ((2 * t2) & 7) + 16 * (t2 >> 2);     // channel e (K dim)
    int m = nt * 16 + (l2 & 15);                    // spatial col
    unsigned lo = (m < HW) ? (unsigned)f2bf(vbc[k * HW + m]) : 0u;
    unsigned hi2 = (m < HW) ? (unsigned)f2bf(vbc[(k + 1) * HW + m]) : 0u;
    ((unsigned*)Bkv)[i] = lo | (hi2 << 16);
  }
  __syncthreads();
  if (wave < 2) {   // out(16x32) = A2(16x512) @ V(512x32)
    int nt = wave;
    v8f acc = {};
    for (int kch = 0; kch < 16; ++kch) {
      v16bf a  = frag_load(&A2[(kch * 32 + lane) * 16]);
      v16bf bf = frag_load(&Bkv[((kch * 2 + nt) * 32 + lane) * 16]);
      acc = wmma_bf16(a, bf, acc);
    }
    float* outc = ws + OFF_OUTC + (size_t)q * DD * HW;
#pragma unroll
    for (int v = 0; v < 8; ++v) {
      int c = c0 + hi * 8 + v;
      int m = nt * 16 + nloc;
      if (m < HW) outc[c * HW + m] = acc[v];
    }
  }
}

// ---------------------------------------------------------------------------
// K6: query output epilogue (fc_o for both branches + residuals).
// ---------------------------------------------------------------------------
__global__ void __launch_bounds__(256)
finalq_kernel(const float* __restrict__ ws, const float* __restrict__ query,
              const float* __restrict__ Wop, const float* __restrict__ bop,
              const float* __restrict__ Woc, const float* __restrict__ boc,
              const float* __restrict__ gp, const float* __restrict__ gc,
              const float* __restrict__ g, float* __restrict__ outq) {
  int idx = blockIdx.x * 256 + threadIdx.x;
  if (idx >= SZ) return;
  int q = idx / (DD * HW);
  int c = (idx / HW) % DD;
  int n = idx % HW;
  // channel branch: fc_o over HW
  const float* ocrow = ws + OFF_OUTC + (size_t)q * DD * HW + c * HW;
  float oc = boc[n];
#pragma unroll
  for (int m = 0; m < HW; ++m) oc += ocrow[m] * Woc[n * HW + m];
  float cout = gc[0] * oc + ws[OFF_VBC + idx];
  // position branch: fc_o over C
  const float* oprow = ws + OFF_OUTP + (size_t)q * HW * DD + n * DD;
  const float* wrow = Wop + (size_t)c * DD;
  float op = bop[c];
  for (int e = 0; e < DD; ++e) op += oprow[e] * wrow[e];
  float pout = gp[0] * op + ws[OFF_VBP + idx];
  outq[idx] = (g[0] * (pout + cout) + query[idx]) * (1.f / 3.f);
}

// ---------------------------------------------------------------------------
// K7: support output (elementwise).
// ---------------------------------------------------------------------------
__global__ void __launch_bounds__(256)
finals_kernel(const float* __restrict__ ws, const float* __restrict__ support,
              const float* __restrict__ beta, float* __restrict__ outs) {
  int idx = blockIdx.x * 256 + threadIdx.x;
  if (idx >= SZ) return;
  outs[idx] =
      (beta[0] * (ws[OFF_VAP + idx] + ws[OFF_VAC + idx]) + support[idx]) *
      (1.f / 3.f);
}

extern "C" void kernel_launch(void* const* d_in, const int* in_sizes, int n_in,
                              void* d_out, int out_size, void* d_ws,
                              size_t ws_size, hipStream_t stream) {
  (void)in_sizes; (void)n_in; (void)out_size; (void)ws_size;
  const float* support = (const float*)d_in[0];
  const float* query   = (const float*)d_in[1];
  const float* Wc      = (const float*)d_in[2];
  const float* Wp      = (const float*)d_in[3];
  const float* Wop     = (const float*)d_in[4];
  const float* bop     = (const float*)d_in[5];
  const float* Woc     = (const float*)d_in[6];
  const float* boc     = (const float*)d_in[7];
  const float* gp      = (const float*)d_in[8];
  const float* gc      = (const float*)d_in[9];
  const float* g       = (const float*)d_in[10];
  const float* bt      = (const float*)d_in[11];
  float* ws  = (float*)d_ws;
  float* out = (float*)d_out;

  qkvc_kernel<<<48, 256, 0, stream>>>(support, query, Wc, ws);
  qkvp_kernel<<<576, 256, 0, stream>>>(support, query, Wp, ws);
  posatt_kernel<<<24, 256, 0, stream>>>(ws);
  outp_kernel<<<24, 256, 0, stream>>>(ws);
  chanatt_kernel<<<768, 256, 0, stream>>>(ws);
  finalq_kernel<<<(SZ + 255) / 256, 256, 0, stream>>>(
      ws, query, Wop, bop, Woc, boc, gp, gc, g, out + SZ);
  finals_kernel<<<(SZ + 255) / 256, 256, 0, stream>>>(ws, support, bt, out);
}
